// FrequencySelfAttention4_35124242547049
// MI455X (gfx1250) — matrix operand order repaired and checked by reading, not compiled, not run on hardware
//
#include <hip/hip_runtime.h>

#define BB 8
#define CC 256
#define NN 4096
#define H0 32
#define W0 32

typedef _Float16 f16;
typedef __attribute__((ext_vector_type(16))) _Float16 v16h;
typedef __attribute__((ext_vector_type(8)))  _Float16 v8h;
typedef __attribute__((ext_vector_type(8)))  float    v8f;

union FragU { v16h v; v8h h[2]; };

// A-matrix fragment (16x32 f16, M x K), memory layout [row][k] row-major.
// Element i of lane L: K = kbase + (i/8)*16 + (L/16)*8 + i%8, M = row (caller adds lane%16).
__device__ __forceinline__ v16h load_a(const f16* base, int row, int rs, int kbase, int lane) {
  const f16* p = base + (size_t)row * rs + kbase + ((lane >> 4) << 3);
  FragU u;
  u.h[0] = *(const v8h*)(p);
  u.h[1] = *(const v8h*)(p + 16);
  return u.v;
}
// B-matrix fragment (32x16 f16, K x N), memory layout [col][k] (K contiguous).
// Element i of lane L: K = kbase + (i/8)*16 + (L/16)*8 + i%8, N = col (caller adds lane%16).
__device__ __forceinline__ v16h load_b(const f16* base, int col, int rs, int kbase, int lane) {
  const f16* p = base + (size_t)col * rs + kbase + ((lane >> 4) << 3);
  FragU u;
  u.h[0] = *(const v8h*)(p);
  u.h[1] = *(const v8h*)(p + 16);
  return u.v;
}

#define WMMA_F16(a, b, c) \
  __builtin_amdgcn_wmma_f32_16x16x32_f16(false, (a), false, (b), (short)0, (c), false, false)

// ---------------- prep kernels ----------------

// bilinear(align_corners) + sigmoid -> scale = 1 + sigmoid(g)
__global__ __launch_bounds__(256) void k_scale(const float* __restrict__ g0,
                                               float* __restrict__ sc) {
  int idx = blockIdx.x * 256 + threadIdx.x;   // B*N
  int b = idx >> 12, n = idx & (NN - 1);
  int ny = n >> 6, nx = n & 63;
  float fy = ny * (31.0f / 63.0f);
  float fx = nx * (31.0f / 63.0f);
  int y0 = (int)fy, x0 = (int)fx;
  float wy = fy - (float)y0, wx = fx - (float)x0;
  int y1 = min(y0 + 1, H0 - 1), x1 = min(x0 + 1, W0 - 1);
  const float* g = g0 + (size_t)b * H0 * W0;
  float v00 = g[y0 * W0 + x0], v01 = g[y0 * W0 + x1];
  float v10 = g[y1 * W0 + x0], v11 = g[y1 * W0 + x1];
  float vy0 = v00 * (1.f - wx) + v01 * wx;
  float vy1 = v10 * (1.f - wx) + v11 * wx;
  float v = vy0 * (1.f - wy) + vy1 * wy;
  sc[idx] = 1.0f + 1.0f / (1.0f + __expf(-v));
}

// convert weights to f16; fold freq_weights into Wq/Wk/Wv columns (refined = fw*x)
__global__ __launch_bounds__(256) void k_wconv(const float* __restrict__ Wq,
                                               const float* __restrict__ Wk,
                                               const float* __restrict__ Wv,
                                               const float* __restrict__ Wo,
                                               const float* __restrict__ fw,
                                               f16* __restrict__ W3,
                                               f16* __restrict__ Wo16) {
  int idx = blockIdx.x * 256 + threadIdx.x;   // 4*256*256
  int m = idx >> 16, r = idx & 65535, ci = r & (CC - 1);
  if (m == 0)      W3[r]          = (f16)(Wq[r] * fw[ci]);
  else if (m == 1) W3[65536 + r]  = (f16)(Wk[r] * fw[ci]);
  else if (m == 2) W3[131072 + r] = (f16)(Wv[r] * fw[ci]);
  else             Wo16[r]        = (f16)Wo[r];
}

// x [b][c][n] f32 -> xT [b][n][c] f16 via LDS 64x64 tile (coalesced both sides)
__global__ __launch_bounds__(256) void k_xt(const float* __restrict__ x,
                                            f16* __restrict__ xT) {
  __shared__ __align__(16) f16 tile[64][72];
  int b = blockIdx.z, c0 = blockIdx.y * 64, n0 = blockIdx.x * 64;
  int tid = threadIdx.x;
  {
    int cl = tid >> 2, nq = (tid & 3) * 16;
    const float4* src = (const float4*)(x + (size_t)b * CC * NN +
                                        (size_t)(c0 + cl) * NN + n0 + nq);
    for (int i = 0; i < 4; ++i) {
      float4 v = src[i];
      tile[cl][nq + 4 * i + 0] = (f16)v.x;
      tile[cl][nq + 4 * i + 1] = (f16)v.y;
      tile[cl][nq + 4 * i + 2] = (f16)v.z;
      tile[cl][nq + 4 * i + 3] = (f16)v.w;
    }
  }
  __syncthreads();
  {
    int nl = tid >> 2, cq = (tid & 3) * 16;
    f16* dst = xT + (size_t)b * NN * CC + (size_t)(n0 + nl) * CC + c0 + cq;
    v8h o0, o1;
    for (int i = 0; i < 8; ++i) { o0[i] = tile[cq + i][nl]; o1[i] = tile[cq + 8 + i][nl]; }
    *(v8h*)dst = o0;
    *(v8h*)(dst + 8) = o1;
  }
}

// ---------------- QKV GEMM ----------------
// q/k stored [b][n][c] (transposed, f16, scaled); v stored [b][c][n] (f16, scaled)
__global__ __launch_bounds__(256) void k_qkv(const f16* __restrict__ xT,
                                             const f16* __restrict__ W3,
                                             const float* __restrict__ bq,
                                             const float* __restrict__ bk,
                                             const float* __restrict__ bv,
                                             const float* __restrict__ scg,
                                             f16* __restrict__ qT,
                                             f16* __restrict__ kT,
                                             f16* __restrict__ v16) {
  int b = blockIdx.y;
  int n_base = blockIdx.x * 64;
  int lane = threadIdx.x & 31, wave = threadIdx.x >> 5;
  int co_wave = wave * 32;
  const f16* xTb = xT + (size_t)b * NN * CC;
  const float* scb = scg + (size_t)b * NN;
  const v8f vzero = {};
  for (int m = 0; m < 3; ++m) {
    const f16* Wm = W3 + (size_t)m * 65536;
    const float* bias = (m == 0) ? bq : ((m == 1) ? bk : bv);
    v8f acc[2][4];
    for (int cb = 0; cb < 2; ++cb)
      for (int nb = 0; nb < 4; ++nb) acc[cb][nb] = vzero;
    for (int kk = 0; kk < CC; kk += 32) {
      v16h a0 = load_a(Wm, co_wave + (lane & 15), CC, kk, lane);
      v16h a1 = load_a(Wm, co_wave + 16 + (lane & 15), CC, kk, lane);
      for (int nb = 0; nb < 4; ++nb) {
        v16h bf = load_b(xTb, n_base + nb * 16 + (lane & 15), CC, kk, lane);
        acc[0][nb] = WMMA_F16(a0, bf, acc[0][nb]);
        acc[1][nb] = WMMA_F16(a1, bf, acc[1][nb]);
      }
    }
    float qmul = (m == 0) ? 0.0625f : 1.0f;  // fold 1/sqrt(C) into q
    for (int cb = 0; cb < 2; ++cb) {
      int co_b = co_wave + cb * 16 + ((lane >> 4) << 3);
      for (int nb = 0; nb < 4; ++nb) {
        int n = n_base + nb * 16 + (lane & 15);
        float s = scb[n] * qmul;
        if (m < 2) {
          v8h pk;
          for (int r = 0; r < 8; ++r)
            pk[r] = (f16)((acc[cb][nb][r] + bias[co_b + r]) * s);
          f16* dst = ((m == 0) ? qT : kT) + (size_t)b * NN * CC + (size_t)n * CC + co_b;
          *(v8h*)dst = pk;
        } else {
          f16* vb = v16 + (size_t)b * CC * NN;
          for (int r = 0; r < 8; ++r)
            vb[(size_t)(co_b + r) * NN + n] = (f16)((acc[cb][nb][r] + bias[co_b + r]) * s);
        }
      }
    }
  }
}

// ---------------- flash attention ----------------
// out[b][n][c] f16 = softmax(q^T k) v ; 64 query rows per workgroup, tiles of 64 keys
__global__ __launch_bounds__(256) void k_attn(const f16* __restrict__ qT,
                                              const f16* __restrict__ kT,
                                              const f16* __restrict__ v16,
                                              f16* __restrict__ aout) {
  __shared__ __align__(16) f16 Qs[64][264];      // staged Q block (padded rows)
  __shared__ __align__(16) float Sbuf[64][68];   // scores f32
  __shared__ __align__(16) f16 Pbuf[64][72];     // softmax numerator f16
  __shared__ float m_run[64], l_run[64], rfac[64];
  __shared__ float pmax[4][64], psum[4][64];
  int b = blockIdx.y;
  int nq = blockIdx.x * 64;
  int tid = threadIdx.x, lane = tid & 31, wave = tid >> 5;
  int row_blk = wave & 3;          // S row block for this wave
  int colp = (wave >> 2) * 2;      // S col blocks colp, colp+1
  int c_wave = wave * 32;          // O channel range for this wave
  int srow = tid & 63, sq = tid >> 6;  // softmax: 4 threads per row, 16 cols each
  const f16* qTb = qT + (size_t)b * NN * CC;
  const f16* kTb = kT + (size_t)b * NN * CC;
  const f16* vb  = v16 + (size_t)b * CC * NN;
  const v8f vzero = {};
  v8f oacc[8];
  for (int i = 0; i < 8; ++i) oacc[i] = vzero;

  // stage Q block: 256 threads, each copies 64 halfs
  {
    int row = tid >> 2, cbase = (tid & 3) * 64;
    for (int i = 0; i < 8; ++i)
      *(v8h*)&Qs[row][cbase + i * 8] =
          *(const v8h*)&qTb[(size_t)(nq + row) * CC + cbase + i * 8];
  }
  if (tid < 64) { m_run[tid] = -1e30f; l_run[tid] = 0.f; }
  __syncthreads();

  for (int t = 0; t < NN / 64; ++t) {
    int mb = t * 64;
    // S = Q^T K for this 64x64 tile (2 subtiles per wave)
    v8f s0 = vzero, s1 = vzero;
    for (int kk = 0; kk < CC; kk += 32) {
      v16h a  = load_a(&Qs[0][0], row_blk * 16 + (lane & 15), 264, kk, lane);
      v16h b0 = load_b(kTb, mb + colp * 16 + (lane & 15), CC, kk, lane);
      v16h b1 = load_b(kTb, mb + (colp + 1) * 16 + (lane & 15), CC, kk, lane);
      s0 = WMMA_F16(a, b0, s0);
      s1 = WMMA_F16(a, b1, s1);
    }
    for (int r = 0; r < 8; ++r) {
      int row = row_blk * 16 + r + ((lane >> 4) << 3);
      Sbuf[row][colp * 16 + (lane & 15)] = s0[r];
      Sbuf[row][(colp + 1) * 16 + (lane & 15)] = s1[r];
    }
    __syncthreads();
    // ---- parallel online softmax: all 256 threads, vectorized ----
    float arr[16];
    {
      const float4* sp = (const float4*)&Sbuf[srow][sq * 16];
      float4 f0 = sp[0], f1 = sp[1], f2 = sp[2], f3 = sp[3];
      ((float4*)arr)[0] = f0; ((float4*)arr)[1] = f1;
      ((float4*)arr)[2] = f2; ((float4*)arr)[3] = f3;
      float mx = fmaxf(fmaxf(fmaxf(f0.x, f0.y), fmaxf(f0.z, f0.w)),
                       fmaxf(fmaxf(fmaxf(f1.x, f1.y), fmaxf(f1.z, f1.w)),
                             fmaxf(fmaxf(fmaxf(f2.x, f2.y), fmaxf(f2.z, f2.w)),
                                   fmaxf(fmaxf(f3.x, f3.y), fmaxf(f3.z, f3.w)))));
      pmax[sq][srow] = mx;
    }
    __syncthreads();
    {
      float m_new = fmaxf(fmaxf(pmax[0][srow], pmax[1][srow]),
                          fmaxf(fmaxf(pmax[2][srow], pmax[3][srow]), m_run[srow]));
      float l = 0.f;
      v8h pk0, pk1;
      for (int i = 0; i < 8; ++i) {
        float p = __expf(arr[i] - m_new);
        pk0[i] = (f16)p; l += p;
      }
      for (int i = 0; i < 8; ++i) {
        float p = __expf(arr[8 + i] - m_new);
        pk1[i] = (f16)p; l += p;
      }
      *(v8h*)&Pbuf[srow][sq * 16] = pk0;
      *(v8h*)&Pbuf[srow][sq * 16 + 8] = pk1;
      psum[sq][srow] = l;
    }
    __syncthreads();
    if (tid < 64) {
      float mold = m_run[tid];
      float m_new = fmaxf(fmaxf(pmax[0][tid], pmax[1][tid]),
                          fmaxf(fmaxf(pmax[2][tid], pmax[3][tid]), mold));
      float rs = __expf(mold - m_new);
      l_run[tid] = l_run[tid] * rs +
                   (psum[0][tid] + psum[1][tid] + psum[2][tid] + psum[3][tid]);
      m_run[tid] = m_new;
      rfac[tid] = rs;
    }
    __syncthreads();
    // rescale O and accumulate O += P * V^T (K = 64 keys)
    for (int nb = 0; nb < 4; ++nb) {
      float f[8];
      for (int r = 0; r < 8; ++r) f[r] = rfac[nb * 16 + r + ((lane >> 4) << 3)];
      for (int cb = 0; cb < 2; ++cb)
        for (int r = 0; r < 8; ++r) oacc[nb * 2 + cb][r] *= f[r];
      for (int kk = 0; kk < 64; kk += 32) {
        v16h pa = load_a(&Pbuf[0][0], nb * 16 + (lane & 15), 72, kk, lane);
        for (int cb = 0; cb < 2; ++cb) {
          v16h vf = load_b(vb, c_wave + cb * 16 + (lane & 15), NN, mb + kk, lane);
          oacc[nb * 2 + cb] = WMMA_F16(pa, vf, oacc[nb * 2 + cb]);
        }
      }
    }
    __syncthreads();
  }
  // epilogue: divide by softmax sum, store [b][n][c] f16
  f16* ab = aout + (size_t)b * NN * CC;
  for (int nb = 0; nb < 4; ++nb)
    for (int cb = 0; cb < 2; ++cb) {
      int c = c_wave + cb * 16 + (lane & 15);
      for (int r = 0; r < 8; ++r) {
        int nloc = nb * 16 + r + ((lane >> 4) << 3);
        float val = oacc[nb * 2 + cb][r] / l_run[nloc];
        ab[(size_t)(nq + nloc) * CC + c] = (f16)val;
      }
    }
}

// ---------------- output projection ----------------
__global__ __launch_bounds__(256) void k_out(const f16* __restrict__ aout,
                                             const f16* __restrict__ Wo16,
                                             const float* __restrict__ bo,
                                             float* __restrict__ out) {
  int b = blockIdx.y;
  int n_base = blockIdx.x * 64;
  int lane = threadIdx.x & 31, wave = threadIdx.x >> 5;
  int co_wave = wave * 32;
  const f16* ab = aout + (size_t)b * NN * CC;
  const v8f vzero = {};
  v8f acc[2][4];
  for (int cb = 0; cb < 2; ++cb)
    for (int nb = 0; nb < 4; ++nb) acc[cb][nb] = vzero;
  for (int kk = 0; kk < CC; kk += 32) {
    v16h a0 = load_a(Wo16, co_wave + (lane & 15), CC, kk, lane);
    v16h a1 = load_a(Wo16, co_wave + 16 + (lane & 15), CC, kk, lane);
    for (int nb = 0; nb < 4; ++nb) {
      v16h bf = load_b(ab, n_base + nb * 16 + (lane & 15), CC, kk, lane);
      acc[0][nb] = WMMA_F16(a0, bf, acc[0][nb]);
      acc[1][nb] = WMMA_F16(a1, bf, acc[1][nb]);
    }
  }
  float* ob = out + (size_t)b * CC * NN;
  for (int cb = 0; cb < 2; ++cb) {
    int co_b = co_wave + cb * 16 + ((lane >> 4) << 3);
    for (int nb = 0; nb < 4; ++nb) {
      int n = n_base + nb * 16 + (lane & 15);
      for (int r = 0; r < 8; ++r)
        ob[(size_t)(co_b + r) * NN + n] = acc[cb][nb][r] + bo[co_b + r];
    }
  }
}

extern "C" void kernel_launch(void* const* d_in, const int* in_sizes, int n_in,
                              void* d_out, int out_size, void* d_ws, size_t ws_size,
                              hipStream_t stream) {
  (void)in_sizes; (void)n_in; (void)out_size; (void)ws_size;
  const float* x  = (const float*)d_in[0];
  const float* g0 = (const float*)d_in[1];
  const float* Wq = (const float*)d_in[2];
  const float* bq = (const float*)d_in[3];
  const float* Wk = (const float*)d_in[4];
  const float* bk = (const float*)d_in[5];
  const float* Wv = (const float*)d_in[6];
  const float* bv = (const float*)d_in[7];
  const float* Wo = (const float*)d_in[8];
  const float* bo = (const float*)d_in[9];
  const float* fw = (const float*)d_in[10];

  char* ws = (char*)d_ws;
  float* scg = (float*)ws; ws += (size_t)BB * NN * sizeof(float);
  f16* W3    = (f16*)ws;   ws += (size_t)3 * CC * CC * sizeof(f16);
  f16* Wo16  = (f16*)ws;   ws += (size_t)CC * CC * sizeof(f16);
  f16* xT    = (f16*)ws;   ws += (size_t)BB * NN * CC * sizeof(f16);
  f16* qT    = (f16*)ws;   ws += (size_t)BB * NN * CC * sizeof(f16);
  f16* kT    = (f16*)ws;   ws += (size_t)BB * NN * CC * sizeof(f16);
  f16* v16   = (f16*)ws;   ws += (size_t)BB * NN * CC * sizeof(f16);
  f16* aout  = (f16*)ws;

  k_scale<<<dim3((BB * NN) / 256), 256, 0, stream>>>(g0, scg);
  k_wconv<<<dim3((4 * CC * CC) / 256), 256, 0, stream>>>(Wq, Wk, Wv, Wo, fw, W3, Wo16);
  k_xt<<<dim3(NN / 64, CC / 64, BB), 256, 0, stream>>>(x, xT);
  k_qkv<<<dim3(NN / 64, BB), 256, 0, stream>>>(xT, W3, bq, bk, bv, scg, qT, kT, v16);
  k_attn<<<dim3(NN / 64, BB), 256, 0, stream>>>(qT, kT, v16, aout);
  k_out<<<dim3(NN / 64, BB), 256, 0, stream>>>(aout, Wo16, bo, (float*)d_out);
}
